// TransformerEncoder_4844723110041
// MI455X (gfx1250) — matrix operand order repaired and checked by reading, not compiled, not run on hardware
//
#include <hip/hip_runtime.h>
#include <math.h>

// Problem constants (match reference)
#define V_ 32000
#define D_ 1024
#define H_ 4096
#define S_ 2048
#define L_ 2
#define B_ 4
#define EPS_ 1e-5f

typedef __bf16 bf16;
typedef __attribute__((ext_vector_type(16))) __bf16 v16bf;
typedef __attribute__((ext_vector_type(8)))  __bf16 bf16x8;
typedef __attribute__((ext_vector_type(8)))  float  v8f;
typedef __attribute__((ext_vector_type(4)))  unsigned int u32x4;
typedef __attribute__((ext_vector_type(8)))  int i32x8;
typedef __attribute__((ext_vector_type(4)))  int i32x4;

// ---------------------------------------------------------------------------
// Weight convert + transpose: in fp32 [K,N] row-major -> out bf16 [N,K]
// ---------------------------------------------------------------------------
__global__ __launch_bounds__(256)
void k_transpose_bf16(const float* __restrict__ in, bf16* __restrict__ out,
                      int K, int N) {
  __shared__ float tile[32][33];
  const int kb = blockIdx.y * 32, nb = blockIdx.x * 32;
  const int tx = threadIdx.x, ty = threadIdx.y;  // block (32, 8)
#pragma unroll
  for (int i = 0; i < 32; i += 8)
    tile[ty + i][tx] = in[(size_t)(kb + ty + i) * N + nb + tx];
  __syncthreads();
#pragma unroll
  for (int i = 0; i < 32; i += 8)
    out[(size_t)(nb + ty + i) * K + kb + tx] = (bf16)tile[tx][ty + i];
}

// ---------------------------------------------------------------------------
// Embedding + sinusoidal positional encoding -> y (fp32) and yb (bf16)
// ---------------------------------------------------------------------------
__global__ __launch_bounds__(256)
void k_embed(const int* __restrict__ x, const float* __restrict__ emb,
             float* __restrict__ y, bf16* __restrict__ yb) {
  const size_t idx = (size_t)blockIdx.x * 256 + threadIdx.x;  // over B*S*D
  const int d = (int)(idx % D_);
  const size_t t = idx / D_;           // flattened (b, s)
  const int s = (int)(t % S_);
  const int tok = x[t];
  const float ang = (float)s * __powf(10000.0f, -(float)d / (float)D_);
  const float pe = ((d & 1) == 0) ? __sinf(ang) : __cosf(ang);
  const float v = emb[(size_t)tok * D_ + d] + pe;
  y[idx] = v;
  yb[idx] = (bf16)v;
}

// ---------------------------------------------------------------------------
// TDM: async DMA of a 2-D bf16 tile [tile_rows x BK] (row length `ld`
// elements) from global into LDS at byte offset lds_off, with hardware
// padding: after every 16 DWORDs (64B = one 32-elem bf16 row) insert
// 4 DWORDs (16B) -> LDS row stride 80B (LSTR=40 bf16).
// D# per CDNA5 ISA ch. 8; groups 2/3 zero => 2-D tensor.
// Toolchain uses the 6-arg builtin form (clang-23 / therock-10.0).
// ---------------------------------------------------------------------------
#define BM 128
#define BN 128
#define BK 32
#define LSTR 40

__device__ __forceinline__
void tdm_load_tile(const bf16* gbase, unsigned lds_off, int ld, int rows_total,
                   int row0, int k0, int tile_rows) {
  const unsigned long long ga =
      (unsigned long long)(uintptr_t)gbase +
      ((unsigned long long)row0 * (unsigned)ld + (unsigned)k0) * 2ull;
  u32x4 g0;
  g0[0] = 1u;                                    // count=1 (valid user D#)
  g0[1] = lds_off;                               // lds_addr (bytes)
  g0[2] = (unsigned)(ga & 0xffffffffu);          // global_addr[31:0]
  g0[3] = (unsigned)((ga >> 32) & 0x01ffffffu)   // global_addr[56:32]
          | (2u << 30);                          // type=2 ("image")
  i32x8 g1;
  g1[0] = (int)((1u << 16)        // data_size = 1 -> 2 bytes
              | (1u << 20)        // pad_enable
              | (3u << 22)        // pad_interval: 16 DWORDs
              | (3u << 25));      // pad_amount: 4 DWORDs
  g1[1] = (int)(((unsigned)ld & 0xffffu) << 16);                 // tensor_dim0 lo16
  g1[2] = (int)((((unsigned)ld >> 16) & 0xffffu)
              | (((unsigned)rows_total & 0xffffu) << 16));       // dim0 hi / dim1 lo
  g1[3] = (int)((((unsigned)rows_total >> 16) & 0xffffu)
              | ((unsigned)BK << 16));                           // dim1 hi / tile_dim0=32
  g1[4] = (int)((unsigned)tile_rows & 0xffffu);                  // tile_dim1; tile_dim2=0
  g1[5] = (int)(unsigned)ld;                                     // tensor_dim0_stride lo32
  g1[6] = 0;                                                     // stride hi16; dim1_stride=0
  g1[7] = 0;
  const i32x4 z4 = {0, 0, 0, 0};
  const i32x8 z8 = {0, 0, 0, 0, 0, 0, 0, 0};
  __builtin_amdgcn_tensor_load_to_lds(g0, g1, z4, z4, z8, 0);
}

// ---------------------------------------------------------------------------
// WMMA bf16 GEMM:  C[M,N] = A[M,K] * Bt[N,K]^T (+ bias[N])
// Double-buffered LDS tiles staged by the Tensor Data Mover (wave 0 issues,
// TENSORcnt + workgroup barrier synchronize). 8 waves, block tile 128x128,
// wave tile 64x32 = 4x2 accumulators, K-step 32.
// ---------------------------------------------------------------------------
template <typename OutT, bool TOUT, bool HASBIAS>
__global__ __launch_bounds__(256)
void k_gemm(const bf16* __restrict__ A, int lda, long long strideA,
            const bf16* __restrict__ Bt, int ldb, long long strideB,
            const float* __restrict__ bias,
            OutT* __restrict__ C, int ldc, long long strideC,
            int M, int N, int K, int SoutRows) {
  __shared__ bf16 sA[2][BM * LSTR];
  __shared__ bf16 sB[2][BN * LSTR];

  const int tid  = threadIdx.x;
  const int lane = tid & 31;
  const int hlf  = lane >> 4;      // lane half 0/1
  const int lhl  = lane & 15;
  const int wm   = (tid >> 5) >> 2;  // 0..1 -> M offset wm*64
  const int wn   = (tid >> 5) & 3;   // 0..3 -> N offset wn*32
  const bool lead = (__builtin_amdgcn_readfirstlane(tid) == 0);  // wave 0 only

  const long long zb = blockIdx.z;
  A  += zb * strideA;
  Bt += zb * strideB;

  const int rowBase = blockIdx.y * BM;
  const int colBase = blockIdx.x * BN;

  const unsigned ldsA[2] = {(unsigned)(uintptr_t)&sA[0][0],
                            (unsigned)(uintptr_t)&sA[1][0]};
  const unsigned ldsB[2] = {(unsigned)(uintptr_t)&sB[0][0],
                            (unsigned)(uintptr_t)&sB[1][0]};

  v8f acc[4][2];
#pragma unroll
  for (int im = 0; im < 4; ++im)
#pragma unroll
    for (int in = 0; in < 2; ++in)
#pragma unroll
      for (int r = 0; r < 8; ++r) acc[im][in][r] = 0.0f;

  // Prologue: DMA first K-tile into buffer 0.
  if (lead) {
    tdm_load_tile(A,  ldsA[0], lda, M, rowBase, 0, BM);
    tdm_load_tile(Bt, ldsB[0], ldb, N, colBase, 0, BN);
  }

  int cur = 0;
  for (int kk = 0; kk < K; kk += BK) {
    const bool has_next = (kk + BK) < K;
    if (lead) {
      if (has_next) {
        // Prefetch next tile into the other buffer, then wait until only the
        // two prefetch ops remain in flight (TENSORcnt decrements in order).
        tdm_load_tile(A,  ldsA[cur ^ 1], lda, M, rowBase, kk + BK, BM);
        tdm_load_tile(Bt, ldsB[cur ^ 1], ldb, N, colBase, kk + BK, BN);
        __builtin_amdgcn_s_wait_tensorcnt(2);
      } else {
        __builtin_amdgcn_s_wait_tensorcnt(0);
      }
    }
    __syncthreads();  // current buffer ready for all waves

    // Fragment loads per ISA 16-bit layouts.
    v16bf afrag[4], bfrag[2];
#pragma unroll
    for (int im = 0; im < 4; ++im) {
      const bf16* ar = &sA[cur][(wm * 64 + im * 16 + lhl) * LSTR];
      union { v16bf v; bf16x8 h[2]; } u;
      u.h[0] = *(const bf16x8*)&ar[hlf * 8];       // K = hlf*8 .. +7
      u.h[1] = *(const bf16x8*)&ar[16 + hlf * 8];  // K = 16+hlf*8 .. +7
      afrag[im] = u.v;
    }
#pragma unroll
    for (int in = 0; in < 2; ++in) {
      const bf16* br = &sB[cur][(wn * 32 + in * 16 + lhl) * LSTR];
      union { v16bf v; bf16x8 h[2]; } u;
      u.h[0] = *(const bf16x8*)&br[hlf * 8];       // K = hlf*8 .. +7
      u.h[1] = *(const bf16x8*)&br[16 + hlf * 8];  // K = 16+hlf*8 .. +7
      bfrag[in] = u.v;
    }

#pragma unroll
    for (int im = 0; im < 4; ++im)
#pragma unroll
      for (int in = 0; in < 2; ++in)
        acc[im][in] = __builtin_amdgcn_wmma_f32_16x16x32_bf16(
            false, afrag[im], false, bfrag[in], (short)0, acc[im][in],
            false, false);

    __syncthreads();  // all waves done reading `cur` before it is re-filled
    cur ^= 1;
  }

  // Bias: one scalar per wave column (hoisted out of the store loops).
  float bw[2] = {0.0f, 0.0f};
  if constexpr (HASBIAS) {
#pragma unroll
    for (int in = 0; in < 2; ++in)
      bw[in] = bias[colBase + wn * 32 + in * 16 + lhl];
  }

  // Epilogue. C/D layout: lane half h, col n = lane&15, VGPR r -> m = h*8+r.
#pragma unroll
  for (int im = 0; im < 4; ++im) {
    const int grow0 = rowBase + wm * 64 + im * 16 + hlf * 8;  // 8 consecutive rows
#pragma unroll
    for (int in = 0; in < 2; ++in) {
      const int gcol = colBase + wn * 32 + in * 16 + lhl;
      if constexpr (TOUT) {
        // Transposed per-batch output: 8 consecutive source rows map to 8
        // contiguous elements of column gcol -> one packed 16B store.
        const long long b = grow0 / SoutRows;
        const int sr = grow0 % SoutRows;
        bf16x8 pack;
#pragma unroll
        for (int r = 0; r < 8; ++r) pack[r] = (bf16)(acc[im][in][r] + bw[in]);
        *(bf16x8*)&C[b * strideC + (size_t)gcol * ldc + sr] = pack;
      } else {
        size_t base = (size_t)(zb * strideC) + (size_t)grow0 * ldc + gcol;
#pragma unroll
        for (int r = 0; r < 8; ++r) {
          C[base] = (OutT)(acc[im][in][r] + bw[in]);
          base += (size_t)ldc;
        }
      }
    }
  }
}

// ---------------------------------------------------------------------------
// Row softmax over scores [B*S, S] (fp32 in) -> probs bf16
// ---------------------------------------------------------------------------
__global__ __launch_bounds__(256)
void k_softmax(const float* __restrict__ scores, bf16* __restrict__ probs) {
  const size_t row = blockIdx.x;
  const float* sr = scores + row * S_;
  bf16* pr = probs + row * S_;
  float vals[8];
  float mx = -1e30f;
#pragma unroll
  for (int i = 0; i < 8; ++i) {
    vals[i] = sr[threadIdx.x + i * 256];
    mx = fmaxf(mx, vals[i]);
  }
  __shared__ float red[256];
  red[threadIdx.x] = mx;
  __syncthreads();
  for (int s = 128; s > 0; s >>= 1) {
    if (threadIdx.x < s) red[threadIdx.x] = fmaxf(red[threadIdx.x], red[threadIdx.x + s]);
    __syncthreads();
  }
  mx = red[0];
  __syncthreads();
  float sum = 0.0f;
#pragma unroll
  for (int i = 0; i < 8; ++i) { vals[i] = __expf(vals[i] - mx); sum += vals[i]; }
  red[threadIdx.x] = sum;
  __syncthreads();
  for (int s = 128; s > 0; s >>= 1) {
    if (threadIdx.x < s) red[threadIdx.x] += red[threadIdx.x + s];
    __syncthreads();
  }
  const float inv = 1.0f / red[0];
#pragma unroll
  for (int i = 0; i < 8; ++i) pr[threadIdx.x + i * 256] = (bf16)(vals[i] * inv);
}

// ---------------------------------------------------------------------------
// LayerNorm over whole (S,D) per sample — 3 deterministic stages.
// ---------------------------------------------------------------------------
__global__ __launch_bounds__(256)
void k_resid_partial(const float* __restrict__ y, const float* __restrict__ delta,
                     float* __restrict__ r, float* __restrict__ partial, int dogelu) {
  const int b = blockIdx.x >> 6;      // sample
  const int chunk = blockIdx.x & 63;  // 64 chunks per sample
  const int CH = (S_ * D_) / 64;
  const size_t base = (size_t)b * (S_ * D_) + (size_t)chunk * CH;
  float s = 0.0f, ss = 0.0f;
  for (int i = threadIdx.x; i < CH; i += 256) {
    float x = delta[base + i];
    if (dogelu) x = 0.5f * x * (1.0f + erff(x * 0.70710678118f));
    const float t = y[base + i] + x;
    r[base + i] = t;
    s += t;
    ss += t * t;
  }
  __shared__ float rs[256], rss[256];
  rs[threadIdx.x] = s;
  rss[threadIdx.x] = ss;
  __syncthreads();
  for (int k = 128; k > 0; k >>= 1) {
    if (threadIdx.x < k) { rs[threadIdx.x] += rs[threadIdx.x + k]; rss[threadIdx.x] += rss[threadIdx.x + k]; }
    __syncthreads();
  }
  if (threadIdx.x == 0) {
    partial[blockIdx.x * 2 + 0] = rs[0];
    partial[blockIdx.x * 2 + 1] = rss[0];
  }
}

__global__ void k_ln_stats(const float* __restrict__ partial, float* __restrict__ stats) {
  const int b = blockIdx.x;   // 64 threads per block
  __shared__ float rs[64], rss[64];
  rs[threadIdx.x]  = partial[(b * 64 + threadIdx.x) * 2 + 0];
  rss[threadIdx.x] = partial[(b * 64 + threadIdx.x) * 2 + 1];
  __syncthreads();
  for (int k = 32; k > 0; k >>= 1) {
    if (threadIdx.x < k) { rs[threadIdx.x] += rs[threadIdx.x + k]; rss[threadIdx.x] += rss[threadIdx.x + k]; }
    __syncthreads();
  }
  if (threadIdx.x == 0) {
    const float n = (float)(S_ * D_);
    const float mean = rs[0] / n;
    const float var = rss[0] / n - mean * mean;
    stats[b * 2 + 0] = mean;
    stats[b * 2 + 1] = rsqrtf(var + EPS_);
  }
}

__global__ __launch_bounds__(256)
void k_ln_apply(const float* __restrict__ r, const float* __restrict__ stats,
                const float* __restrict__ w, const float* __restrict__ bsh,
                float* __restrict__ yout, bf16* __restrict__ yb) {
  const size_t idx = (size_t)blockIdx.x * 256 + threadIdx.x;  // B*S*D
  const size_t per = (size_t)S_ * D_;
  const int b = (int)(idx / per);
  const size_t o = idx % per;
  const float mean = stats[b * 2 + 0], rstd = stats[b * 2 + 1];
  const float v = (r[idx] - mean) * rstd * w[o] + bsh[o];
  if (yout) yout[idx] = v;
  yb[idx] = (bf16)v;
}

// ---------------------------------------------------------------------------
// Host launcher
// ---------------------------------------------------------------------------
extern "C" void kernel_launch(void* const* d_in, const int* in_sizes, int n_in,
                              void* d_out, int out_size, void* d_ws, size_t ws_size,
                              hipStream_t stream) {
  (void)in_sizes; (void)n_in; (void)out_size; (void)ws_size;
  const int*   x    = (const int*)d_in[0];
  const float* emb  = (const float*)d_in[1];
  const float* wq   = (const float*)d_in[2];
  const float* bq   = (const float*)d_in[3];
  const float* wk   = (const float*)d_in[4];
  const float* bk   = (const float*)d_in[5];
  const float* wv   = (const float*)d_in[6];
  const float* bv   = (const float*)d_in[7];
  const float* wo   = (const float*)d_in[8];
  const float* bo   = (const float*)d_in[9];
  const float* w1   = (const float*)d_in[10];
  const float* b1   = (const float*)d_in[11];
  const float* w2   = (const float*)d_in[12];
  const float* b2   = (const float*)d_in[13];
  const float* ln1w = (const float*)d_in[14];
  const float* ln1b = (const float*)d_in[15];
  const float* ln2w = (const float*)d_in[16];
  const float* ln2b = (const float*)d_in[17];
  float* out = (float*)d_out;

  // Workspace layout (256B aligned carve-outs).
  char* p = (char*)d_ws;
  auto carve = [&](size_t bytes) {
    void* q = (void*)p;
    p += (bytes + 255) & ~(size_t)255;
    return q;
  };
  const size_t BSD = (size_t)B_ * S_ * D_;      // 8M
  const size_t BSH = (size_t)B_ * S_ * H_;      // 32M
  const size_t BSS = (size_t)B_ * S_ * S_;      // 16M
  const size_t WSZ = (size_t)D_ * H_;           // 4M per weight matrix

  bf16*  yb     = (bf16*) carve(BSD * 2);
  bf16*  wT     = (bf16*) carve((size_t)L_ * 6 * WSZ * 2);
  bf16*  qb     = (bf16*) carve(BSH * 2);
  bf16*  kb     = (bf16*) carve(BSH * 2);
  bf16*  vT     = (bf16*) carve(BSH * 2);       // per batch [H][S]
  bf16*  attb   = (bf16*) carve(BSH * 2);
  bf16*  hb     = (bf16*) carve(BSH * 2);
  bf16*  probs  = (bf16*) carve(BSS * 2);
  float* scores = (float*)carve(BSS * 4);
  float* proj   = (float*)carve(BSD * 4);
  float* rbuf   = (float*)carve(BSD * 4);
  float* ybuf   = (float*)carve(BSD * 4);
  float* partial= (float*)carve((size_t)B_ * 64 * 2 * 4);
  float* stats  = (float*)carve((size_t)B_ * 2 * 4);

  const dim3 tb(32, 8);
  // Weight convert+transpose: order per layer: wqT wkT wvT w1T woT w2T
  for (int l = 0; l < L_; ++l) {
    bf16* base = wT + (size_t)l * 6 * WSZ;
    // D x H matrices -> [H][D]
    k_transpose_bf16<<<dim3(H_ / 32, D_ / 32), tb, 0, stream>>>(wq + (size_t)l * WSZ, base + 0 * WSZ, D_, H_);
    k_transpose_bf16<<<dim3(H_ / 32, D_ / 32), tb, 0, stream>>>(wk + (size_t)l * WSZ, base + 1 * WSZ, D_, H_);
    k_transpose_bf16<<<dim3(H_ / 32, D_ / 32), tb, 0, stream>>>(wv + (size_t)l * WSZ, base + 2 * WSZ, D_, H_);
    k_transpose_bf16<<<dim3(H_ / 32, D_ / 32), tb, 0, stream>>>(w1 + (size_t)l * WSZ, base + 3 * WSZ, D_, H_);
    // H x D matrices -> [D][H]
    k_transpose_bf16<<<dim3(D_ / 32, H_ / 32), tb, 0, stream>>>(wo + (size_t)l * WSZ, base + 4 * WSZ, H_, D_);
    k_transpose_bf16<<<dim3(D_ / 32, H_ / 32), tb, 0, stream>>>(w2 + (size_t)l * WSZ, base + 5 * WSZ, H_, D_);
  }

  // Embedding + positional encoding.
  k_embed<<<(unsigned)(BSD / 256), 256, 0, stream>>>(x, emb, ybuf, yb);

  const int M = B_ * S_;  // 8192 flattened rows
  for (int l = 0; l < L_; ++l) {
    bf16* wqT = wT + ((size_t)l * 6 + 0) * WSZ;
    bf16* wkT = wT + ((size_t)l * 6 + 1) * WSZ;
    bf16* wvT = wT + ((size_t)l * 6 + 2) * WSZ;
    bf16* w1T = wT + ((size_t)l * 6 + 3) * WSZ;
    bf16* woT = wT + ((size_t)l * 6 + 4) * WSZ;
    bf16* w2T = wT + ((size_t)l * 6 + 5) * WSZ;

    // q, k: [M,H] bf16
    k_gemm<bf16, false, true><<<dim3(H_ / BN, M / BM, 1), 256, 0, stream>>>(
        yb, D_, 0, wqT, D_, 0, bq + (size_t)l * H_, qb, H_, 0, M, H_, D_, 0);
    k_gemm<bf16, false, true><<<dim3(H_ / BN, M / BM, 1), 256, 0, stream>>>(
        yb, D_, 0, wkT, D_, 0, bk + (size_t)l * H_, kb, H_, 0, M, H_, D_, 0);
    // v written transposed per batch: vT[b][h][s]
    k_gemm<bf16, true, true><<<dim3(H_ / BN, M / BM, 1), 256, 0, stream>>>(
        yb, D_, 0, wvT, D_, 0, bv + (size_t)l * H_, vT, S_, (long long)H_ * S_,
        M, H_, D_, S_);

    // scores[b] = q[b] (S,H) x k[b]^T -> fp32 (unscaled, as reference)
    k_gemm<float, false, false><<<dim3(S_ / BN, S_ / BM, B_), 256, 0, stream>>>(
        qb, H_, (long long)S_ * H_, kb, H_, (long long)S_ * H_, nullptr,
        scores, S_, (long long)S_ * S_, S_, S_, H_, 0);

    // row softmax -> bf16 probs
    k_softmax<<<(unsigned)(B_ * S_), 256, 0, stream>>>(scores, probs);

    // att[b] = probs[b] (S,S) x v[b] (S,H)  via Bt = vT[b] ([H][S])
    k_gemm<bf16, false, false><<<dim3(H_ / BN, S_ / BM, B_), 256, 0, stream>>>(
        probs, S_, (long long)S_ * S_, vT, S_, (long long)H_ * S_, nullptr,
        attb, H_, (long long)S_ * H_, S_, H_, S_, 0);

    // output projection: [M,D] fp32 (+bo)
    k_gemm<float, false, true><<<dim3(D_ / BN, M / BM, 1), 256, 0, stream>>>(
        attb, H_, 0, woT, H_, 0, bo + (size_t)l * D_, proj, D_, 0, M, D_, H_, 0);

    // LN1 over (S,D) per sample
    k_resid_partial<<<B_ * 64, 256, 0, stream>>>(ybuf, proj, rbuf, partial, 0);
    k_ln_stats<<<B_, 64, 0, stream>>>(partial, stats);
    k_ln_apply<<<(unsigned)(BSD / 256), 256, 0, stream>>>(
        rbuf, stats, ln1w + (size_t)l * S_ * D_, ln1b + (size_t)l * S_ * D_,
        ybuf, yb);

    // FFN: h = y*w1+b1 (bf16), ffn = gelu(h*w2+b2)
    k_gemm<bf16, false, true><<<dim3(H_ / BN, M / BM, 1), 256, 0, stream>>>(
        yb, D_, 0, w1T, D_, 0, b1 + (size_t)l * H_, hb, H_, 0, M, H_, D_, 0);
    k_gemm<float, false, true><<<dim3(D_ / BN, M / BM, 1), 256, 0, stream>>>(
        hb, H_, 0, w2T, H_, 0, b2 + (size_t)l * D_, proj, D_, 0, M, D_, H_, 0);

    // LN2 (gelu applied to proj inside residual stage)
    k_resid_partial<<<B_ * 64, 256, 0, stream>>>(ybuf, proj, rbuf, partial, 1);
    k_ln_stats<<<B_, 64, 0, stream>>>(partial, stats);
    const bool last = (l == L_ - 1);
    k_ln_apply<<<(unsigned)(BSD / 256), 256, 0, stream>>>(
        rbuf, stats, ln2w + (size_t)l * S_ * D_, ln2b + (size_t)l * S_ * D_,
        last ? out : ybuf, yb);
  }
}
